// LocalGConvLSTMCell_75522704932944
// MI455X (gfx1250) — compile-verified, matrix operand order checked by reading, not run on hardware
//
#include <hip/hip_runtime.h>
#include <hip/hip_bf16.h>

typedef __attribute__((ext_vector_type(16))) __bf16 v16bf;
typedef __attribute__((ext_vector_type(8)))  __bf16 v8bf;
typedef __attribute__((ext_vector_type(8)))  float  v8f;

static __device__ __forceinline__ unsigned short f2bf(float f) {
    unsigned int u = __float_as_uint(f);
    u += 0x7FFFu + ((u >> 16) & 1u);          // round-to-nearest-even
    return (unsigned short)(u >> 16);
}

static __device__ __forceinline__ void atomAddF(float* p, float v) {
    unsafeAtomicAdd(p, v);                    // hardware global_atomic_add_f32
}

// ---------------- zero scratch ----------------
__global__ __launch_bounds__(256)
void zero_kernel(float4* __restrict__ p, long long n4) {
    long long i = (long long)blockIdx.x * 256 + threadIdx.x;
    if (i < n4) p[i] = make_float4(0.f, 0.f, 0.f, 0.f);
}

// ---------------- degree: deg[src] += w ----------------
__global__ __launch_bounds__(256)
void deg_kernel(const long long* __restrict__ ei, const float* __restrict__ ew,
                float* __restrict__ deg, long long E) {
    long long e = (long long)blockIdx.x * 256 + threadIdx.x;
    if (e < E) atomAddF(&deg[ei[e]], ew[e]);
}

// ---------------- dis = deg>0 ? rsqrt(deg) : 0 (in place) ----------------
__global__ __launch_bounds__(256)
void dis_kernel(float* __restrict__ deg, int N) {
    int n = blockIdx.x * 256 + threadIdx.x;
    if (n < N) {
        float d = deg[n];
        deg[n] = (d > 0.f) ? rsqrtf(d) : 0.f;
    }
}

// ---------------- edge scatter: Lx[dst] += norm*x[src], Lh[dst] += norm*h[src] ----
// one wave32 per edge; each lane covers 4 of 128 channels
__global__ __launch_bounds__(256)
void scatter_kernel(const long long* __restrict__ ei, const float* __restrict__ ew,
                    const float* __restrict__ dis,
                    const float* __restrict__ x, const float* __restrict__ h,
                    float* __restrict__ Lx, float* __restrict__ Lh, long long E) {
    long long e = (long long)blockIdx.x * 8 + (threadIdx.x >> 5);
    if (e >= E) return;
    const int lane = threadIdx.x & 31;
    const long long s = ei[e];
    const long long d = ei[E + e];
    const float nrm = -(dis[s] * ew[e] * dis[d]);

    const float4 vx = ((const float4*)(x + s * 128))[lane];
    const float4 vh = ((const float4*)(h + s * 128))[lane];
    float* lx = Lx + d * 128 + lane * 4;
    float* lh = Lh + d * 128 + lane * 4;
    atomAddF(lx + 0, nrm * vx.x); atomAddF(lx + 1, nrm * vx.y);
    atomAddF(lx + 2, nrm * vx.z); atomAddF(lx + 3, nrm * vx.w);
    atomAddF(lh + 0, nrm * vh.x); atomAddF(lh + 1, nrm * vh.y);
    atomAddF(lh + 2, nrm * vh.z); atomAddF(lh + 3, nrm * vh.w);
}

// ---------------- pack concat weights [512K x 512N] into bf16 WMMA B fragments ----
// frag f = nt*16 + kb : 32 lanes x 16 bf16, lane chunk contiguous (32B)
// B lane layout (mirror of ISA A table): lane<16 -> N=lane, K {0..7,16..23};
// lane>=16 -> N=lane-16, K {8..15,24..31}
__global__ __launch_bounds__(256)
void bpack_kernel(const float* __restrict__ Wx0, const float* __restrict__ Wx1,
                  const float* __restrict__ Wh0, const float* __restrict__ Wh1,
                  unsigned short* __restrict__ bpack) {
    int idx = blockIdx.x * 256 + threadIdx.x;      // 0 .. 262143
    if (idx >= 32 * 16 * 32 * 16) return;
    int e    = idx & 15;
    int lane = (idx >> 4) & 31;
    int frag = idx >> 9;
    int kb   = frag & 15;
    int nt   = frag >> 4;
    int n = nt * 16 + (lane & 15);
    int k = kb * 32 + ((e >= 8) ? 16 : 0) + ((lane >= 16) ? 8 : 0) + (e & 7);
    int g = n >> 7, j = n & 127;
    int c = k & 127, sel = k >> 7;
    const float* W = (sel == 0) ? Wx0 : (sel == 1) ? Wx1 : (sel == 2) ? Wh0 : Wh1;
    bpack[idx] = f2bf(W[g * 16384 + c * 128 + j]);
}

// ---------------- fused WMMA GEMM (16 x 512 per block) + LSTM elementwise ----------
__global__ __launch_bounds__(256)
void gemm_lstm_kernel(const float* __restrict__ x,  const float* __restrict__ Lx,
                      const float* __restrict__ h,  const float* __restrict__ Lh,
                      const unsigned short* __restrict__ bpack,
                      const float* __restrict__ bx, const float* __restrict__ bh,
                      const float* __restrict__ c_prev,
                      float* __restrict__ outH, float* __restrict__ outC, int N) {
    __shared__ __attribute__((aligned(16))) unsigned short ldsA[16 * 512]; // 16 KB bf16
    __shared__ float ldsC[16 * 512];                                       // 32 KB f32

    const int tid = threadIdx.x;
    const int mbase = blockIdx.x * 16;

    // Stage A panel: rows = 16 nodes, K = concat[x | Lx | h | Lh] (512), as bf16
    const float* srcs[4] = {x, Lx, h, Lh};
    for (int idx = tid; idx < 16 * 512; idx += 256) {
        int row = idx >> 9, k = idx & 511;
        int node = mbase + row;
        float v = 0.f;
        if (node < N) v = srcs[k >> 7][(long long)node * 128 + (k & 127)];
        ldsA[idx] = f2bf(v);
    }
    __syncthreads();

    const int wave  = tid >> 5;
    const int lane  = tid & 31;
    const int rowA  = lane & 15;
    const int khalf = (lane >> 4) * 8;    // 0 or 8

    v8f zero = {0.f, 0.f, 0.f, 0.f, 0.f, 0.f, 0.f, 0.f};
    v8f acc[4];
#pragma unroll
    for (int t = 0; t < 4; ++t) acc[t] = zero;

    const v16bf* B = (const v16bf*)bpack;
#pragma unroll 2
    for (int kb = 0; kb < 16; ++kb) {
        const unsigned short* arow = &ldsA[rowA * 512 + kb * 32];
        v8bf alo = *(const v8bf*)(arow + khalf);          // K {0..7} or {8..15}
        v8bf ahi = *(const v8bf*)(arow + 16 + khalf);     // K {16..23} or {24..31}
        v16bf a = __builtin_shufflevector(alo, ahi, 0, 1, 2, 3, 4, 5, 6, 7,
                                          8, 9, 10, 11, 12, 13, 14, 15);
#pragma unroll
        for (int t = 0; t < 4; ++t) {
            int nt = wave * 4 + t;
            v16bf b = B[(nt * 16 + kb) * 32 + lane];
            acc[t] = __builtin_amdgcn_wmma_f32_16x16x32_bf16(
                false, a, false, b, (short)0, acc[t], false, false);
        }
    }

    // C layout: lane<16 -> N=lane, M=v ; lane>=16 -> N=lane-16, M=v+8
#pragma unroll
    for (int t = 0; t < 4; ++t) {
        int ncol = (wave * 4 + t) * 16 + (lane & 15);
        int mrow = 8 * (lane >> 4);
#pragma unroll
        for (int v = 0; v < 8; ++v)
            ldsC[(mrow + v) * 512 + ncol] = acc[t][v];
    }
    __syncthreads();

    // LSTM elementwise: gates order i, f, g, o at cols j, 128+j, 256+j, 384+j
    for (int q = tid; q < 16 * 128; q += 256) {
        int row = q >> 7, j = q & 127;
        int node = mbase + row;
        if (node >= N) continue;
        float gi = ldsC[row * 512 +       j] + bx[      j] + bh[      j];
        float gf = ldsC[row * 512 + 128 + j] + bx[128 + j] + bh[128 + j];
        float gg = ldsC[row * 512 + 256 + j] + bx[256 + j] + bh[256 + j];
        float go = ldsC[row * 512 + 384 + j] + bx[384 + j] + bh[384 + j];
        float i_ = 1.f / (1.f + __expf(-gi));
        float f_ = 1.f / (1.f + __expf(-gf));
        float g_ = tanhf(gg);
        float o_ = 1.f / (1.f + __expf(-go));
        float cc = f_ * c_prev[(long long)node * 128 + j] + i_ * g_;
        float hh = o_ * tanhf(cc);
        outH[(long long)node * 128 + j] = hh;
        outC[(long long)node * 128 + j] = cc;
    }
}

extern "C" void kernel_launch(void* const* d_in, const int* in_sizes, int n_in,
                              void* d_out, int out_size, void* d_ws, size_t ws_size,
                              hipStream_t stream) {
    const float*     x   = (const float*)d_in[0];
    const long long* ei  = (const long long*)d_in[1];
    const float*     ew  = (const float*)d_in[2];
    const float*     h   = (const float*)d_in[3];
    const float*     c   = (const float*)d_in[4];
    const float*     Wx0 = (const float*)d_in[5];
    const float*     Wx1 = (const float*)d_in[6];
    const float*     Wh0 = (const float*)d_in[7];
    const float*     Wh1 = (const float*)d_in[8];
    const float*     bx  = (const float*)d_in[9];
    const float*     bh  = (const float*)d_in[10];

    const int       N = in_sizes[0] / 128;
    const long long E = in_sizes[2];

    char* ws = (char*)d_ws;
    unsigned short* bpack = (unsigned short*)ws;                     // 512 KB
    const size_t degOff   = 524288;
    const size_t degBytes = ((size_t)N * 4 + 255) & ~(size_t)255;
    const size_t LBytes   = (size_t)N * 128 * 4;
    float* deg = (float*)(ws + degOff);
    float* Lx  = (float*)(ws + degOff + degBytes);
    float* Lh  = (float*)(ws + degOff + degBytes + LBytes);

    float* outH = (float*)d_out;
    float* outC = outH + (size_t)N * 128;

    // 1) zero deg + Lx + Lh (contiguous region)
    const size_t zero4 = (degBytes + 2 * LBytes) / 16;
    zero_kernel<<<(unsigned)((zero4 + 255) / 256), 256, 0, stream>>>(
        (float4*)(ws + degOff), (long long)zero4);

    // 2) weight pre-pack into WMMA B-fragment bf16 layout (independent)
    bpack_kernel<<<1024, 256, 0, stream>>>(Wx0, Wx1, Wh0, Wh1, bpack);

    // 3) degree accumulation
    deg_kernel<<<(unsigned)((E + 255) / 256), 256, 0, stream>>>(ei, ew, deg, E);

    // 4) deg -> deg^{-1/2} in place
    dis_kernel<<<(N + 255) / 256, 256, 0, stream>>>(deg, N);

    // 5) edge scatter (one wave per edge)
    scatter_kernel<<<(unsigned)((E + 7) / 8), 256, 0, stream>>>(
        ei, ew, deg, x, h, Lx, Lh, E);

    // 6) fused WMMA GEMM + LSTM
    gemm_lstm_kernel<<<(N + 15) / 16, 256, 0, stream>>>(
        x, Lx, h, Lh, bpack, bx, bh, c, outH, outC, N);
}
